// SecondOrderPredictor_24421184045157
// MI455X (gfx1250) — matrix operand order repaired and checked by reading, not compiled, not run on hardware
//
#include <hip/hip_runtime.h>
#include <hip/hip_bf16.h>
#include <math.h>

// ---------------- problem constants (match reference) ----------------
#define MODES 16
#define CIN   64
#define COUT  64
#define BATCH 4
#define GRID  1024
#define DXF   (1.0f / 1024.0f)
#define HORIZON_SCALE 0.5f   // DT_STAR * CAUSAL_SAFETY
#define FILTER_STRENGTH 4.0f
#define POLE_OFFSET_SCALE 0.1f
#define POLE_MIN 0.1f
#define POLE_MAX 100.0f

typedef __attribute__((ext_vector_type(16))) __bf16 v16bf;
typedef __attribute__((ext_vector_type(8)))  __bf16 v8bf;
typedef __attribute__((ext_vector_type(8)))  float  v8f;

// ---------------- workspace layout (bytes) ----------------
// xm   : 64 f32
// poles: 16 f32
// EfM  : 4*16*1024 bf16   ([b][k][d] = mask_b(d) * filt_k * exp(-p_k d dx))
// Weff : 16*64*64 bf16    ([k][o][c])
// xT   : 4*1024*64 bf16   ([b][j][c])
// z    : 16*4*64*1024 bf16([k][b][o][j])
#define OFF_XM     0
#define OFF_POLES  1024
#define OFF_EFM    4096
#define OFF_WEFF   (OFF_EFM  + BATCH*MODES*GRID*2)      // 4096 + 131072
#define OFF_XT     (OFF_WEFF + MODES*COUT*CIN*2)        // + 131072
#define OFF_Z      (OFF_XT   + BATCH*GRID*CIN*2)        // + 524288

// =====================================================================
//  WMMA fragment helpers (wave32, A/B in contiguous k order k = 16*half + s)
// =====================================================================

// A-matrix 16x32 bf16 from row-major [row][col] matrix (ld in elements).
// lanes 0-15: M=lane, K in {0..15}; lanes 16-31: M=lane-16, K in {16..31}
__device__ inline v16bf load_a_tile(const __bf16* base, int ld, int row0, int col0, int lane) {
    int half = lane >> 4;                 // 0 or 1
    int r = row0 + (lane & 15);
    const __bf16* p0 = base + (long)r * ld + col0 + 16 * half;  // 32B contiguous
    union { v16bf v; v8bf h[2]; } u;
    u.h[0] = *(const v8bf*)p0;            // K = 16*half + 0 .. +7
    u.h[1] = *(const v8bf*)(p0 + 8);      // K = 16*half + 8 .. +15
    return u.v;
}

// B-matrix 32x16 bf16 where memory holds B transposed: row-major [n][k].
// lane = N (+16 selects K-half): lanes 0-15 hold K=0..15, lanes 16-31 hold K=16..31
__device__ inline v16bf load_b_tileT(const __bf16* baseT, int ld, int k0, int n0, int lane) {
    int half = lane >> 4;
    int n = n0 + (lane & 15);
    const __bf16* p = baseT + (long)n * ld + k0 + 16 * half;    // 32B contiguous
    union { v16bf v; v8bf h[2]; } u;
    u.h[0] = *(const v8bf*)p;
    u.h[1] = *(const v8bf*)(p + 8);
    return u.v;
}

// Synthesize the (pre-masked) Toeplitz kernel tile as a B fragment from a
// bf16 LDS table: B[kk, n] = Ek[| (ibase+n) - (jbase+kk) |]
__device__ inline v16bf make_kern_tile(const __bf16* __restrict__ Ek,
                                       int ibase, int jbase, int lane) {
    int half = lane >> 4;
    int base = ibase + (lane & 15) - jbase - 16 * half;   // d = |base - e|
    v16bf r;
#pragma unroll
    for (int e = 0; e < 16; ++e) {
        int d = base - e; d = d < 0 ? -d : d;
        r[e] = Ek[d];                                     // ds_load_u16 gather
    }
    return r;
}

// C/D f32 16x16: VGPR r -> M=r (lanes 0-15) / M=r+8 (lanes 16-31), N=lane%16
__device__ inline void store_c_f32(float* base, int ld, int row0, int col0, v8f c, int lane) {
    int half = lane >> 4;
    int n = col0 + (lane & 15);
#pragma unroll
    for (int r = 0; r < 8; ++r)
        base[(long)(row0 + r + 8 * half) * ld + n] = c[r];
}

__device__ inline void store_c_bf16(__bf16* base, int ld, int row0, int col0, v8f c, int lane) {
    int half = lane >> 4;
    int n = col0 + (lane & 15);
#pragma unroll
    for (int r = 0; r < 8; ++r)
        base[(long)(row0 + r + 8 * half) * ld + n] = (__bf16)c[r];
}

// =====================================================================
//  Prep kernels
// =====================================================================

// x_mean[c] = mean over (b, g) of x[b][c][g]
__global__ __launch_bounds__(256) void mean_kernel(const float* __restrict__ x,
                                                   float* __restrict__ xm) {
    int c = blockIdx.x;
    float s = 0.f;
    for (int idx = threadIdx.x; idx < BATCH * GRID; idx += 256) {
        int b = idx >> 10, g = idx & (GRID - 1);
        s += x[((long)b * CIN + c) * GRID + g];
    }
    __shared__ float red[256];
    red[threadIdx.x] = s;
    __syncthreads();
    for (int o = 128; o > 0; o >>= 1) {
        if (threadIdx.x < o) red[threadIdx.x] += red[threadIdx.x + o];
        __syncthreads();
    }
    if (threadIdx.x == 0) xm[c] = red[0] * (1.0f / (BATCH * GRID));
}

// poles[k] = clip(softplus(log_poles + 0.1*tanh(xm @ pole_w + pole_b)), 0.1, 100)
__global__ __launch_bounds__(32) void poles_kernel(const float* __restrict__ xm,
                                                   const float* __restrict__ pole_w,
                                                   const float* __restrict__ pole_b,
                                                   const float* __restrict__ log_poles,
                                                   float* __restrict__ poles) {
    int k = threadIdx.x;
    if (k >= MODES) return;
    float acc = pole_b[k];
    for (int c = 0; c < CIN; ++c) acc += xm[c] * pole_w[c * MODES + k];
    float off = tanhf(acc);
    float zv = log_poles[k] + POLE_OFFSET_SCALE * off;
    float p = log1pf(expf(zv));           // softplus
    p = fminf(fmaxf(p, POLE_MIN), POLE_MAX);
    poles[k] = p;
}

// EfM[b][k][d] = (d*dx <= c*_b*0.5) ? exp(-4*(k/15)^2) * exp(-poles[k]*d*dx) : 0
// (mask applied at f32 precision, stored bf16)
__global__ __launch_bounds__(256) void efm_kernel(const float* __restrict__ poles,
                                                  const float* __restrict__ c_star,
                                                  __bf16* __restrict__ EfM) {
    int idx = blockIdx.x * 256 + threadIdx.x;      // 65536 total
    int b = idx >> 14;
    int k = (idx >> 10) & (MODES - 1);
    int d = idx & (GRID - 1);
    float kn = (float)k * (1.0f / (MODES - 1));
    float filt = expf(-FILTER_STRENGTH * kn * kn);
    float horizon = c_star[b] * HORIZON_SCALE;
    float dist = (float)d * DXF;
    float v = (dist <= horizon) ? filt * expf(-poles[k] * dist) : 0.0f;
    EfM[idx] = (__bf16)v;
}

// Weff[k][o][c] = sigmoid(log_amp[k][c][o]) * cos(phase[k][c][o])   (bf16, transposed)
__global__ __launch_bounds__(256) void weff_kernel(const float* __restrict__ wla,
                                                   const float* __restrict__ wph,
                                                   __bf16* __restrict__ Weff) {
    int idx = blockIdx.x * 256 + threadIdx.x;      // 65536 total
    int k = idx >> 12;
    int rem = idx & 4095;
    int o = rem >> 6, c = rem & 63;
    long src = (long)k * CIN * COUT + (long)c * COUT + o;
    float amp = 1.0f / (1.0f + expf(-wla[src]));
    float w = amp * cosf(wph[src]);
    Weff[(long)k * COUT * CIN + (long)o * CIN + c] = (__bf16)w;
}

// xT[b][j][c] = x[b][c][j]  (bf16)
__global__ __launch_bounds__(256) void xt_kernel(const float* __restrict__ x,
                                                 __bf16* __restrict__ xT) {
    int idx = blockIdx.x * 256 + threadIdx.x;      // 262144 total
    int b = idx >> 16;
    int rem = idx & 65535;
    int j = rem >> 6, c = rem & 63;
    xT[idx] = (__bf16)x[((long)b * CIN + c) * GRID + j];
}

// =====================================================================
//  GEMM 1: z[k][b][o][j] = sum_c Weff[k][o][c] * xT[b][j][c]
//  One wave per (k,b,ot, group of 4 j-tiles): A fragments loaded once,
//  reused for 4 tiles -> 8 WMMAs / wave.
// =====================================================================
__global__ __launch_bounds__(256) void zgemm_kernel(const __bf16* __restrict__ Weff,
                                                    const __bf16* __restrict__ xT,
                                                    __bf16* __restrict__ z) {
    int wave = (blockIdx.x * 256 + threadIdx.x) >> 5;   // 4096 waves
    int lane = threadIdx.x & 31;
    int jg = wave & 15;  wave >>= 4;     // 16 groups of 4 j-tiles
    int ot = wave & 3;   wave >>= 2;     // 4  o-tiles
    int b  = wave & 3;
    int k  = wave >> 2;                  // 16 modes

    const __bf16* W = Weff + (long)k * COUT * CIN;   // [o][c]
    const __bf16* X = xT + (long)b * GRID * CIN;     // [j][c]
    __bf16* Z = z + ((long)(k * BATCH + b) * COUT) * GRID;   // [o][j]

    v16bf a0 = load_a_tile(W, CIN, ot * 16, 0,  lane);   // resident A fragments
    v16bf a1 = load_a_tile(W, CIN, ot * 16, 32, lane);

#pragma unroll
    for (int t = 0; t < 4; ++t) {
        int jt = jg * 4 + t;
        v16bf b0 = load_b_tileT(X, CIN, 0,  jt * 16, lane);
        v16bf b1 = load_b_tileT(X, CIN, 32, jt * 16, lane);
        v8f acc = {};
        acc = __builtin_amdgcn_wmma_f32_16x16x32_bf16(false, a0, false, b0, (short)0, acc, false, false);
        acc = __builtin_amdgcn_wmma_f32_16x16x32_bf16(false, a1, false, b1, (short)0, acc, false, false);
        store_c_bf16(Z, GRID, ot * 16, jt * 16, acc, lane);
    }
}

// =====================================================================
//  GEMM 2 (banded, Toeplitz kernel): one wave owns ALL 4 o-tiles of a
//  (b, i-tile) so every synthesized kern tile feeds 4 WMMAs.
//  Pre-masked bf16 Ef table (32 KB for this block's b) staged into LDS.
// =====================================================================
__global__ __launch_bounds__(256) void outgemm_kernel(const __bf16* __restrict__ z,
                                                      const __bf16* __restrict__ EfM,
                                                      const float* __restrict__ c_star,
                                                      float* __restrict__ out) {
    __shared__ __bf16 sEfM[MODES * GRID];            // 32 KB of the 320 KB WGP LDS
    int b = blockIdx.x >> 3;                         // 8 waves/block, 64 i-tiles per b
    {
        const unsigned* src = (const unsigned*)(EfM + (long)b * MODES * GRID);
        unsigned* dst = (unsigned*)sEfM;
        for (int i = threadIdx.x; i < MODES * GRID / 2; i += 256) dst[i] = src[i];
    }
    __syncthreads();

    int wave = (blockIdx.x * 256 + threadIdx.x) >> 5;   // 256 waves
    int lane = threadIdx.x & 31;
    int it = wave & 63;                  // 64 i-tiles (b = wave >> 6 == blockIdx.x>>3)

    float horizon = c_star[b] * HORIZON_SCALE;
    int R = (int)ceilf(horizon * (float)GRID) + 1;
    int ibase = it * 16;
    int jlo = ibase - R;       if (jlo < 0) jlo = 0;
    jlo &= ~31;                                       // 32-aligned tile start
    int jhi = ibase + 15 + R;  if (jhi > GRID - 1) jhi = GRID - 1;

    v8f acc[4] = {{}, {}, {}, {}};
    for (int k = 0; k < MODES; ++k) {
        const __bf16* Ek = sEfM + k * GRID;
        const __bf16* Zk = z + ((long)(k * BATCH + b) * COUT) * GRID;  // [o][j]
        if (k + 1 < MODES) {            // warm L2/L0 for next mode's band rows
            const __bf16* Zn = z + ((long)((k + 1) * BATCH + b) * COUT) * GRID;
            __builtin_prefetch(Zn + (long)(lane & 15) * GRID + jlo, 0, 3);
            __builtin_prefetch(Zn + (long)(32 + (lane & 15)) * GRID + jlo, 0, 3);
        }
        for (int jb = jlo; jb <= jhi; jb += 32) {
            v16bf kb = make_kern_tile(Ek, ibase, jb, lane);   // shared by 4 WMMAs
#pragma unroll
            for (int ot = 0; ot < 4; ++ot) {
                v16bf a = load_a_tile(Zk, GRID, ot * 16, jb, lane);
                acc[ot] = __builtin_amdgcn_wmma_f32_16x16x32_bf16(
                    false, a, false, kb, (short)0, acc[ot], false, false);
            }
        }
    }
#pragma unroll
    for (int ot = 0; ot < 4; ++ot)
        store_c_f32(out + (long)b * COUT * GRID, GRID, ot * 16, ibase, acc[ot], lane);
}

// =====================================================================
//  Host launcher
// =====================================================================
extern "C" void kernel_launch(void* const* d_in, const int* in_sizes, int n_in,
                              void* d_out, int out_size, void* d_ws, size_t ws_size,
                              hipStream_t stream) {
    (void)in_sizes; (void)n_in; (void)out_size; (void)ws_size;

    const float* x         = (const float*)d_in[0];
    const float* wla       = (const float*)d_in[1];
    const float* wph       = (const float*)d_in[2];
    const float* log_poles = (const float*)d_in[3];
    const float* pole_w    = (const float*)d_in[4];
    const float* pole_b    = (const float*)d_in[5];
    const float* c_star    = (const float*)d_in[6];
    float* out = (float*)d_out;

    char* ws = (char*)d_ws;
    float*  xm    = (float*)(ws + OFF_XM);
    float*  poles = (float*)(ws + OFF_POLES);
    __bf16* EfM   = (__bf16*)(ws + OFF_EFM);
    __bf16* Weff  = (__bf16*)(ws + OFF_WEFF);
    __bf16* xT    = (__bf16*)(ws + OFF_XT);
    __bf16* z     = (__bf16*)(ws + OFF_Z);

    mean_kernel <<<CIN,  256, 0, stream>>>(x, xm);
    poles_kernel<<<1,    32,  0, stream>>>(xm, pole_w, pole_b, log_poles, poles);
    efm_kernel  <<<(BATCH * MODES * GRID) / 256, 256, 0, stream>>>(poles, c_star, EfM);
    weff_kernel <<<(MODES * CIN * COUT) / 256, 256, 0, stream>>>(wla, wph, Weff);
    xt_kernel   <<<(BATCH * CIN * GRID) / 256, 256, 0, stream>>>(x, xT);

    // 4096 tile-waves (8 WMMA each), 8 waves / 256-thread block
    zgemm_kernel  <<<(MODES * BATCH * 4 * 16) / 8, 256, 0, stream>>>(Weff, xT, z);
    // 256 waves, each owning 4 o-tiles of one (b, i-tile)
    outgemm_kernel<<<(BATCH * 64) / 8, 256, 0, stream>>>(z, EfM, c_star, out);
}
